// RecursiveDecoder_88192858456378
// MI455X (gfx1250) — matrix operand order repaired and checked by reading, not compiled, not run on hardware
//
#include <hip/hip_runtime.h>
#include <hip/hip_bf16.h>

typedef __attribute__((ext_vector_type(16))) __bf16 v16bf;
typedef __attribute__((ext_vector_type(8)))  __bf16 v8bf;
typedef __attribute__((ext_vector_type(8)))  float  v8f;

static __device__ __forceinline__ v8f wmma_bf16(v16bf a, v16bf b, v8f c) {
    // D = A(16x32) * B(32x16) + C, fp32 accumulate
    return __builtin_amdgcn_wmma_f32_16x16x32_bf16(false, a, false, b, (short)0, c, false, false);
}

// ---------------- prep: fp32 (K x N) -> bf16 B^T (Npad x K) ----------------
__global__ void trans_bf16_k(const float* __restrict__ src, int K, int N, int ld,
                             int Npad, __bf16* __restrict__ dst) {
    int idx = blockIdx.x * 256 + threadIdx.x;
    if (idx >= Npad * K) return;
    int n = idx / K, k = idx - n * K;
    float v = (n < N) ? src[k * ld + n] : 0.0f;
    dst[idx] = (__bf16)v;
}

// fp32 -> bf16 straight copy with zero pad (for W_ee which is already T-major)
__global__ void cvt_pad_bf16_k(const float* __restrict__ src, __bf16* __restrict__ dst,
                               int count, int srcCount) {
    int i = blockIdx.x * 256 + threadIdx.x;
    if (i < count) dst[i] = (i < srcCount) ? (__bf16)src[i] : (__bf16)0.0f;
}

// ---------------- stage 0: pf = relu(parent @ W_parent + b) ----------------
__global__ __launch_bounds__(256) void parent_gemv_k(
    const float* __restrict__ pf, const float* __restrict__ Wp,
    const float* __restrict__ bp, float* __restrict__ CF0, __bf16* __restrict__ CFb) {
    __shared__ float sp[256];
    int tid = threadIdx.x;
    sp[tid] = pf[tid];
    __syncthreads();
    int j = blockIdx.x * 256 + tid;
    float acc = bp[j];
#pragma unroll 4
    for (int k = 0; k < 256; ++k) acc = fmaf(sp[k], Wp[k * 65536 + j], acc);
    acc = fmaxf(acc, 0.0f);
    CF0[j] = acc;
    int m = j >> 8, h = j & 255;
    CFb[m * 768 + h] = (__bf16)acc;   // iter-0 slice of feats_cat
}

// ---------------- child exists logits + node_exists + hasEdge=0 -------------
__global__ void exists_k(const float* __restrict__ CF0, const float* __restrict__ Wex,
                         const float* __restrict__ bex, float* __restrict__ outCE,
                         int* __restrict__ ne, int* __restrict__ hasEdge) {
    int m = threadIdx.x;
    float acc = bex[0];
    for (int h = 0; h < 256; ++h) acc = fmaf(CF0[m * 256 + h], Wex[h], acc);
    outCE[m] = acc;
    ne[m] = acc > 0.0f ? 1 : 0;
    if (m == 0) *hasEdge = 0;
}

// ---------------- generic bf16 WMMA GEMM: C = act(A @ B^T + bias) ----------
__global__ __launch_bounds__(256) void wmma_gemm_k(
    const __bf16* __restrict__ A, int lda,
    const __bf16* __restrict__ BT, int ldb,
    const float* __restrict__ bias,
    float* __restrict__ C, int ldc,
    __bf16* __restrict__ Cb, int ldcb,
    int Kt, int Ntiles, int Nreal, int doRelu) {
    int wave = threadIdx.x >> 5, lane = threadIdx.x & 31;
    int tile = blockIdx.x * 8 + wave;
    if (tile >= 16 * Ntiles) return;
    int mt = tile / Ntiles, nt = tile - mt * Ntiles;
    int r16 = lane & 15;
    int kbA = (lane < 16) ? 0 : 8;
    int kbB = (lane < 16) ? 0 : 16;
    const __bf16* Arow = A + (mt * 16 + r16) * lda;
    const __bf16* Bcol = BT + (nt * 16 + r16) * ldb;
    v8f acc = {0.f, 0.f, 0.f, 0.f, 0.f, 0.f, 0.f, 0.f};
    for (int kk = 0; kk < Kt; ++kk) {
        v8bf lo = *(const v8bf*)(Arow + kk * 32 + kbA);
        v8bf hi = *(const v8bf*)(Arow + kk * 32 + kbA + 16);
        v16bf af;
#pragma unroll
        for (int e = 0; e < 8; ++e) { af[e] = lo[e]; af[e + 8] = hi[e]; }
        v8bf blo = *(const v8bf*)(Bcol + kk * 32 + kbA);
        v8bf bhi = *(const v8bf*)(Bcol + kk * 32 + kbA + 16);
        v16bf bfv;
#pragma unroll
        for (int e = 0; e < 8; ++e) { bfv[e] = blo[e]; bfv[e + 8] = bhi[e]; }
        acc = wmma_bf16(af, bfv, acc);
    }
    int c = lane & 15, col = nt * 16 + c;
    if (col < Nreal) {
        float bb = bias ? bias[col] : 0.0f;
#pragma unroll
        for (int j = 0; j < 8; ++j) {
            int row = mt * 16 + j + ((lane >> 4) << 3);
            float v = acc[j] + bb;
            if (doRelu) v = fmaxf(v, 0.0f);
            C[row * ldc + col] = v;
            if (Cb) Cb[row * ldcb + col] = (__bf16)v;
        }
    }
}

// ------------- edge pass: eel[m,n,t] = EL[m,n,:]·Wee[t,:]+bee, + has_edge ---
__global__ __launch_bounds__(256) void edge_k(
    const float* __restrict__ aE, const float* __restrict__ bE,
    const float* __restrict__ bel, const __bf16* __restrict__ BTee,
    const float* __restrict__ bee, const int* __restrict__ ne,
    int* __restrict__ hasEdge, float* __restrict__ outEE) {
    int m = blockIdx.x, tid = threadIdx.x, lane = tid & 31, wave = tid >> 5;
    __shared__ float sA[256];
    sA[tid] = aE[m * 256 + tid] + bel[tid];
    __syncthreads();
    int nem = ne[m];
    int r16 = lane & 15;
    int kbA = (lane < 16) ? 0 : 8;
    int kbB = (lane < 16) ? 0 : 16;
    unsigned flag = 0;
#pragma unroll
    for (int q = 0; q < 2; ++q) {
        int nt = wave * 2 + q;
        int n = nt * 16 + r16;
        const float* brow = bE + n * 256;
        v8f acc = {0.f, 0.f, 0.f, 0.f, 0.f, 0.f, 0.f, 0.f};
        for (int kk = 0; kk < 8; ++kk) {
            v16bf af;
#pragma unroll
            for (int e = 0; e < 8; ++e) {
                int h0 = kk * 32 + kbA + e;
                af[e]     = (__bf16)fmaxf(sA[h0] + brow[h0], 0.0f);
                af[e + 8] = (__bf16)fmaxf(sA[h0 + 16] + brow[h0 + 16], 0.0f);
            }
            v8bf blo = *(const v8bf*)(BTee + r16 * 256 + kk * 32 + kbA);
            v8bf bhi = *(const v8bf*)(BTee + r16 * 256 + kk * 32 + kbA + 16);
            v16bf bfv;
#pragma unroll
            for (int e = 0; e < 8; ++e) { bfv[e] = blo[e]; bfv[e + 8] = bhi[e]; }
            acc = wmma_bf16(af, bfv, acc);
        }
        int c = lane & 15;
        if (c < 4) {
            float bb = bee[c];
#pragma unroll
            for (int j = 0; j < 8; ++j) {
                int n2 = nt * 16 + j + ((lane >> 4) << 3);
                float v = acc[j] + bb;
                outEE[(m * 256 + n2) * 4 + c] = v;
                if (v > 0.0f && nem && ne[n2]) flag = 1;
            }
        }
    }
    if (__ballot(flag)) { if (lane == 0) atomicOr(hasEdge, 1); }
}

// ------------- fused message passing: new_cf[m,h]=max_{n,t} masked relu -----
__global__ __launch_bounds__(256) void msg_k(
    const float* __restrict__ aE, const float* __restrict__ bE,
    const float* __restrict__ bel,
    const float* __restrict__ u, const float* __restrict__ bne,
    const float* __restrict__ v, const __bf16* __restrict__ BTwe,
    const float* __restrict__ eel, const float* __restrict__ Wt,
    const int* __restrict__ ne, const int* __restrict__ hasEdgeP,
    const float* __restrict__ cfPrev, float* __restrict__ cfNext,
    __bf16* __restrict__ cfbNext) {
    int m = blockIdx.x, tid = threadIdx.x, lane = tid & 31, wave = tid >> 5;
    __shared__ float sA[256];        // a[m]+b_el
    __shared__ float sU[256];        // u[m]+b_ne
    __shared__ float sWt[1024];      // Wt[t][h]
    __shared__ float sEel[1024];     // eel[m][n][t]
    __shared__ int   sNe[256];
    __shared__ __bf16 sEL[16 * 264]; // one n-tile of edge_latents, bf16, padded
    sA[tid] = aE[m * 256 + tid] + bel[tid];
    sU[tid] = u[m * 256 + tid] + bne[tid];
    sNe[tid] = ne[tid];
#pragma unroll
    for (int t = 0; t < 4; ++t) {
        sWt[t * 256 + tid] = Wt[t * 256 + tid];
        sEel[tid * 4 + t]  = eel[(m * 256 + tid) * 4 + t];
    }
    int nem = ne[m];
    int hasE = *hasEdgeP;
    int r16 = lane & 15;
    int kbA = (lane < 16) ? 0 : 8;
    int kbB = (lane < 16) ? 0 : 16;
    int c = lane & 15;
    float mx[2][8];
#pragma unroll
    for (int q = 0; q < 2; ++q)
#pragma unroll
        for (int j = 0; j < 8; ++j) mx[q][j] = 0.0f;

    for (int nt = 0; nt < 16; ++nt) {
        __syncthreads();
        {   // build bf16 EL tile for rows n = nt*16 .. nt*16+15
            int row = tid >> 4, cb = tid & 15;
            const float* brow = bE + (nt * 16 + row) * 256;
#pragma unroll
            for (int e = 0; e < 16; ++e) {
                int h = cb * 16 + e;
                sEL[row * 264 + h] = (__bf16)fmaxf(sA[h] + brow[h], 0.0f);
            }
        }
        __syncthreads();
#pragma unroll
        for (int q = 0; q < 2; ++q) {
            int ht = wave * 2 + q;
            v8f acc = {0.f, 0.f, 0.f, 0.f, 0.f, 0.f, 0.f, 0.f};
#pragma unroll
            for (int kk = 0; kk < 8; ++kk) {
                v8bf lo = *(const v8bf*)(&sEL[r16 * 264 + kk * 32 + kbA]);
                v8bf hi = *(const v8bf*)(&sEL[r16 * 264 + kk * 32 + kbA + 16]);
                v16bf af;
#pragma unroll
                for (int e = 0; e < 8; ++e) { af[e] = lo[e]; af[e + 8] = hi[e]; }
                v8bf blo = *(const v8bf*)(BTwe + (ht * 16 + c) * 256 + kk * 32 + kbA);
                v8bf bhi = *(const v8bf*)(BTwe + (ht * 16 + c) * 256 + kk * 32 + kbA + 16);
                v16bf bfv;
#pragma unroll
                for (int e = 0; e < 8; ++e) { bfv[e] = blo[e]; bfv[e + 8] = bhi[e]; }
                acc = wmma_bf16(af, bfv, acc);
            }
            int h = ht * 16 + c;
            float uv = sU[h];
#pragma unroll
            for (int j = 0; j < 8; ++j) {
                int n = nt * 16 + j + ((lane >> 4) << 3);
                float base = acc[j] + uv + v[n * 256 + h];
                int ok = nem & sNe[n];
                float mv = mx[q][j];
#pragma unroll
                for (int t = 0; t < 4; ++t) {
                    float e = sEel[n * 4 + t];
                    float pre = fmaf(e, sWt[t * 256 + h], base);
                    float msg = ((e > 0.0f) && ok) ? fmaxf(pre, 0.0f) : 0.0f;
                    mv = fmaxf(mv, msg);
                }
                mx[q][j] = mv;
            }
        }
    }
#pragma unroll
    for (int q = 0; q < 2; ++q) {
        float r = mx[q][0];
#pragma unroll
        for (int j = 1; j < 8; ++j) r = fmaxf(r, mx[q][j]);
        r = fmaxf(r, __shfl_xor(r, 16, 32));
        if (lane < 16) {
            int h = (wave * 2 + q) * 16 + lane;
            float val = hasE ? r : cfPrev[m * 256 + h];
            cfNext[m * 256 + h] = val;
            cfbNext[m * 768 + h] = (__bf16)val;
        }
    }
}

extern "C" void kernel_launch(void* const* d_in, const int* in_sizes, int n_in,
                              void* d_out, int out_size, void* d_ws, size_t ws_size,
                              hipStream_t stream) {
    (void)in_sizes; (void)n_in; (void)out_size; (void)ws_size;
    const float* pf   = (const float*)d_in[0];
    const float* Wp   = (const float*)d_in[1];
    const float* bp   = (const float*)d_in[2];
    const float* Wex  = (const float*)d_in[3];
    const float* bex  = (const float*)d_in[4];
    const float* Wel  = (const float*)d_in[5];
    const float* belv = (const float*)d_in[6];
    const float* Wee  = (const float*)d_in[7];
    const float* bee  = (const float*)d_in[8];
    const float* Wne  = (const float*)d_in[9];
    const float* bnev = (const float*)d_in[10];
    const float* Wch  = (const float*)d_in[11];
    const float* bch  = (const float*)d_in[12];
    const float* Wsem = (const float*)d_in[13];
    const float* bsem = (const float*)d_in[14];
    const float* Wch2 = (const float*)d_in[15];
    const float* bch2 = (const float*)d_in[16];

    float* out    = (float*)d_out;
    float* outOCF = out;                  // (1,256,256)
    float* outSEM = out + 65536;          // (1,256,40)
    float* outCE  = out + 75776;          // (1,256,1)
    float* outEE  = out + 76032;          // (1,256,256,4)

    char* w = (char*)d_ws;
    size_t off = 0;
    auto alloc = [&](size_t bytes) -> char* {
        char* p = w + off; off = (off + bytes + 255) & ~(size_t)255; return p;
    };
    float*  CF0   = (float*)alloc(65536 * 4);
    float*  CF1   = (float*)alloc(65536 * 4);
    float*  CF2   = (float*)alloc(65536 * 4);
    float*  aE    = (float*)alloc(65536 * 4);
    float*  bEm   = (float*)alloc(65536 * 4);
    float*  u     = (float*)alloc(65536 * 4);
    float*  vv    = (float*)alloc(65536 * 4);
    float*  hid   = (float*)alloc(65536 * 4);
    __bf16* CFb   = (__bf16*)alloc(256 * 768 * 2);  // [CF0|CF1|CF2] bf16
    __bf16* hidb  = (__bf16*)alloc(65536 * 2);
    __bf16* BTeli = (__bf16*)alloc(65536 * 2);
    __bf16* BTelj = (__bf16*)alloc(65536 * 2);
    __bf16* BTwi0 = (__bf16*)alloc(65536 * 2);
    __bf16* BTwj0 = (__bf16*)alloc(65536 * 2);
    __bf16* BTwe0 = (__bf16*)alloc(65536 * 2);
    __bf16* BTwi1 = (__bf16*)alloc(65536 * 2);
    __bf16* BTwj1 = (__bf16*)alloc(65536 * 2);
    __bf16* BTwe1 = (__bf16*)alloc(65536 * 2);
    __bf16* BTch  = (__bf16*)alloc(256 * 768 * 2);
    __bf16* BTsem = (__bf16*)alloc(48 * 256 * 2);
    __bf16* BTch2 = (__bf16*)alloc(65536 * 2);
    __bf16* BTee  = (__bf16*)alloc(16 * 256 * 2);
    int*    ne    = (int*)alloc(256 * 4);
    int*    hasE  = (int*)alloc(256);

    // weight prep (bf16 B^T layouts)
    trans_bf16_k<<<256, 256, 0, stream>>>(Wel,           256, 256, 256, 256, BTeli);
    trans_bf16_k<<<256, 256, 0, stream>>>(Wel + 65536,   256, 256, 256, 256, BTelj);
    trans_bf16_k<<<256, 256, 0, stream>>>(Wne,                     256, 256, 256, 256, BTwi0);
    trans_bf16_k<<<256, 256, 0, stream>>>(Wne + 256 * 256,         256, 256, 256, 256, BTwj0);
    trans_bf16_k<<<256, 256, 0, stream>>>(Wne + 512 * 256,         256, 256, 256, 256, BTwe0);
    trans_bf16_k<<<256, 256, 0, stream>>>(Wne + (772 + 0) * 256,   256, 256, 256, 256, BTwi1);
    trans_bf16_k<<<256, 256, 0, stream>>>(Wne + (772 + 256) * 256, 256, 256, 256, 256, BTwj1);
    trans_bf16_k<<<256, 256, 0, stream>>>(Wne + (772 + 512) * 256, 256, 256, 256, 256, BTwe1);
    trans_bf16_k<<<768, 256, 0, stream>>>(Wch,  768, 256, 256, 256, BTch);
    trans_bf16_k<<<48,  256, 0, stream>>>(Wsem, 256, 40,  40,  48,  BTsem);
    trans_bf16_k<<<256, 256, 0, stream>>>(Wch2, 256, 256, 256, 256, BTch2);
    cvt_pad_bf16_k<<<16, 256, 0, stream>>>(Wee, BTee, 16 * 256, 4 * 256);

    // stage 0 + exists
    parent_gemv_k<<<256, 256, 0, stream>>>(pf, Wp, bp, CF0, CFb);
    exists_k<<<1, 256, 0, stream>>>(CF0, Wex, bex, outCE, ne, hasE);

    // edge latent projections a = cf@W_el_i, b = cf@W_el_j
    wmma_gemm_k<<<32, 256, 0, stream>>>(CFb, 768, BTeli, 256, nullptr, aE,  256, nullptr, 0, 8, 16, 256, 0);
    wmma_gemm_k<<<32, 256, 0, stream>>>(CFb, 768, BTelj, 256, nullptr, bEm, 256, nullptr, 0, 8, 16, 256, 0);

    // edge exists logits + has_edge
    edge_k<<<256, 256, 0, stream>>>(aE, bEm, belv, BTee, bee, ne, hasE, outEE);

    // two message-passing iterations
    const __bf16* BTwi[2] = {BTwi0, BTwi1};
    const __bf16* BTwj[2] = {BTwj0, BTwj1};
    const __bf16* BTwe[2] = {BTwe0, BTwe1};
    for (int i = 0; i < 2; ++i) {
        wmma_gemm_k<<<32, 256, 0, stream>>>(CFb + 256 * i, 768, BTwi[i], 256, nullptr, u,  256, nullptr, 0, 8, 16, 256, 0);
        wmma_gemm_k<<<32, 256, 0, stream>>>(CFb + 256 * i, 768, BTwj[i], 256, nullptr, vv, 256, nullptr, 0, 8, 16, 256, 0);
        msg_k<<<256, 256, 0, stream>>>(aE, bEm, belv, u, bnev + 256 * i, vv, BTwe[i],
                                       outEE, Wne + (i * 772 + 768) * 256, ne, hasE,
                                       (i == 0) ? CF0 : CF1,
                                       (i == 0) ? CF1 : CF2,
                                       CFb + 256 * (i + 1));
    }

    // heads: hidden = relu(feats_cat @ W_child + b), sem, child2
    wmma_gemm_k<<<32, 256, 0, stream>>>(CFb,  768, BTch,  768, bch,  hid,    256, hidb, 256, 24, 16, 256, 1);
    wmma_gemm_k<<<6,  256, 0, stream>>>(hidb, 256, BTsem, 256, bsem, outSEM, 40,  nullptr, 0, 8, 3,  40,  0);
    wmma_gemm_k<<<32, 256, 0, stream>>>(hidb, 256, BTch2, 256, bch2, outOCF, 256, nullptr, 0, 8, 16, 256, 1);
}
